// down_block_26792005992604
// MI455X (gfx1250) — matrix operand order repaired and checked by reading, not compiled, not run on hardware
//
#include <hip/hip_runtime.h>

typedef __attribute__((ext_vector_type(16))) _Float16 v16h;
typedef __attribute__((ext_vector_type(8)))  _Float16 v8h;
typedef __attribute__((ext_vector_type(8)))  float    v8f;

#define B_SZ   16
#define V_IN   40962
#define V_POOL 10242
#define C_IN   64
#define C_OUT  128
#define EPS_BN 1e-5f

// ---------------------------------------------------------------------------
// 1) Mean-pool gather: x (B, C_IN, V_IN) f32 -> h0 (B, V_POOL, C_IN) f16
// ---------------------------------------------------------------------------
__global__ void pool_kernel(const float* __restrict__ x,
                            const int* __restrict__ pidx,
                            _Float16* __restrict__ h0) {
    size_t i = (size_t)blockIdx.x * blockDim.x + threadIdx.x;
    const size_t total = (size_t)B_SZ * V_POOL * C_IN;
    if (i >= total) return;
    int c = (int)(i & (C_IN - 1));
    size_t bp = i >> 6;                 // (b * V_POOL + p)
    int p = (int)(bp % V_POOL);
    int b = (int)(bp / V_POOL);
    const float* xc = x + ((size_t)b * C_IN + c) * V_IN;
    float s = 0.f;
#pragma unroll
    for (int k = 0; k < 7; ++k) {
        int v = pidx[7 * p + k];
        s += xc[v];
    }
    h0[i] = (_Float16)(s * (1.0f / 7.0f));
}

// ---------------------------------------------------------------------------
// 2) Weight convert+transpose: W (K, 128) f32 -> WT (128, K) f16
// ---------------------------------------------------------------------------
__global__ void wconv_kernel(const float* __restrict__ W,
                             _Float16* __restrict__ WT, int K) {
    int i = blockIdx.x * blockDim.x + threadIdx.x;
    if (i >= K * C_OUT) return;
    int n = i / K;
    int k = i - n * K;
    WT[i] = (_Float16)W[(size_t)k * C_OUT + n];
}

// ---------------------------------------------------------------------------
// 3) Gathered GEMM via WMMA f16->f32.
//    A row r=(b,p): K dim = 7*CIN laid out as [neighbor n][channel c].
//    One wave handles a 16-row M tile x all 128 output channels (8 N-tiles).
//    A-frag (16x32 f16): lane = row (lane&15); chunks of 8 halves at local
//      k offsets 16*hi and 16*hi+8 (hi = lane>>4) -> contiguous in source row.
//    B-frag (32x16 f16): lane = column (lane&15); 16 consecutive K starting
//      at 16*hi -> contiguous in WT (K-transposed weights).
// ---------------------------------------------------------------------------
template <int CIN>
__global__ void __launch_bounds__(256)
gemm_kernel(const _Float16* __restrict__ A,      // (B*V_POOL, CIN) f16
            const int* __restrict__ nidx,        // 7 * V_POOL
            const _Float16* __restrict__ WT,     // (128, 7*CIN) f16
            const float* __restrict__ bias,      // 128
            float* __restrict__ Cout) {          // (B*V_POOL, 128) f32
    const int KDIM = 7 * CIN;
    const int M_TILES = (B_SZ * V_POOL) / 16;    // 10242
    int wave = threadIdx.x >> 5;
    int lane = threadIdx.x & 31;
    int rowTile = blockIdx.x * 8 + wave;
    if (rowTile >= M_TILES) return;              // wave-uniform: EXEC stays all-1s

    int rbase = rowTile * 16;
    int lrow  = lane & 15;
    int hi    = lane >> 4;
    int r = rbase + lrow;
    int p = r % V_POOL;
    int b = r / V_POOL;
    const int* ni = nidx + 7 * p;

    v8f acc[8] = {};

#pragma unroll
    for (int n = 0; n < 7; ++n) {
        int v = ni[n];
        const _Float16* arow = A + ((size_t)b * V_POOL + v) * CIN;
#pragma unroll
        for (int ks = 0; ks < CIN / 32; ++ks) {
            v8h lo = *(const v8h*)(arow + ks * 32 + 16 * hi);
            v8h hh = *(const v8h*)(arow + ks * 32 + 16 * hi + 8);
            v16h afrag;
#pragma unroll
            for (int e = 0; e < 8; ++e) { afrag[e] = lo[e]; afrag[8 + e] = hh[e]; }
            int gk = n * CIN + ks * 32 + 16 * hi;
#pragma unroll
            for (int t = 0; t < 8; ++t) {
                const _Float16* bptr = WT + (size_t)(t * 16 + lrow) * KDIM + gk;
                v16h bfrag = *(const v16h*)bptr;
                acc[t] = __builtin_amdgcn_wmma_f32_16x16x32_f16(
                    /*neg_a=*/false, afrag, /*neg_b=*/false, bfrag,
                    /*c_mod=*/(short)0, acc[t], /*reuse_a=*/false, /*reuse_b=*/false);
            }
        }
    }

    // C/D layout: VGPR j -> M = j + 8*hi ; lane -> N = t*16 + (lane&15)
#pragma unroll
    for (int t = 0; t < 8; ++t) {
        int col = t * 16 + lrow;
        float bv = bias[col];
#pragma unroll
        for (int j = 0; j < 8; ++j) {
            int row = rbase + j + 8 * hi;
            Cout[(size_t)row * C_OUT + col] = acc[t][j] + bv;
        }
    }
}

// ---------------------------------------------------------------------------
// 4) Per-(b,channel) BN stats over the vertex axis (biased variance)
// ---------------------------------------------------------------------------
__global__ void bnstats_kernel(const float* __restrict__ Cm,
                               float* __restrict__ mean,
                               float* __restrict__ rinv) {
    __shared__ float ss[256];
    __shared__ float sq[256];
    int bn = blockIdx.x;                 // b*128 + n
    int b = bn >> 7, n = bn & 127;
    const float* base = Cm + (size_t)b * V_POOL * C_OUT + n;
    float s = 0.f, s2 = 0.f;
    for (int pp = threadIdx.x; pp < V_POOL; pp += 256) {
        float v = base[(size_t)pp * C_OUT];
        s += v; s2 += v * v;
    }
    ss[threadIdx.x] = s; sq[threadIdx.x] = s2;
    __syncthreads();
    for (int st = 128; st > 0; st >>= 1) {
        if (threadIdx.x < st) {
            ss[threadIdx.x] += ss[threadIdx.x + st];
            sq[threadIdx.x] += sq[threadIdx.x + st];
        }
        __syncthreads();
    }
    if (threadIdx.x == 0) {
        float m = ss[0] * (1.0f / V_POOL);
        float var = sq[0] * (1.0f / V_POOL) - m * m;
        mean[bn] = m;
        rinv[bn] = rsqrtf(var + EPS_BN);
    }
}

// ---------------------------------------------------------------------------
// 5) BN-apply + LeakyReLU -> f16 (conv2 input)
// ---------------------------------------------------------------------------
__global__ void bnact_f16_kernel(const float* __restrict__ Cm,
                                 const float* __restrict__ mean,
                                 const float* __restrict__ rinv,
                                 const float* __restrict__ g,
                                 const float* __restrict__ beta,
                                 _Float16* __restrict__ out) {
    size_t i = (size_t)blockIdx.x * blockDim.x + threadIdx.x;
    if (i >= (size_t)B_SZ * V_POOL * C_OUT) return;
    int n = (int)(i & (C_OUT - 1));
    size_t bp = i >> 7;
    int b = (int)(bp / V_POOL);
    int bn = b * C_OUT + n;
    float v = (Cm[i] - mean[bn]) * rinv[bn] * g[n] + beta[n];
    v = (v >= 0.f) ? v : 0.2f * v;
    out[i] = (_Float16)v;
}

// ---------------------------------------------------------------------------
// 6) BN-apply + LeakyReLU + transpose -> out (B, 128, V_POOL) f32
// ---------------------------------------------------------------------------
__global__ void bnact_out_kernel(const float* __restrict__ Cm,
                                 const float* __restrict__ mean,
                                 const float* __restrict__ rinv,
                                 const float* __restrict__ g,
                                 const float* __restrict__ beta,
                                 float* __restrict__ out) {
    size_t i = (size_t)blockIdx.x * blockDim.x + threadIdx.x;
    if (i >= (size_t)B_SZ * V_POOL * C_OUT) return;
    int n = (int)(i & (C_OUT - 1));
    size_t bp = i >> 7;
    int p = (int)(bp % V_POOL);
    int b = (int)(bp / V_POOL);
    int bn = b * C_OUT + n;
    float v = (Cm[i] - mean[bn]) * rinv[bn] * g[n] + beta[n];
    v = (v >= 0.f) ? v : 0.2f * v;
    out[((size_t)b * C_OUT + n) * V_POOL + p] = v;
}

// ---------------------------------------------------------------------------
extern "C" void kernel_launch(void* const* d_in, const int* in_sizes, int n_in,
                              void* d_out, int out_size, void* d_ws, size_t ws_size,
                              hipStream_t stream) {
    const float* x    = (const float*)d_in[0];
    const int*   pidx = (const int*)d_in[1];
    const int*   nidx = (const int*)d_in[2];
    const float* W1   = (const float*)d_in[3];
    const float* b1   = (const float*)d_in[4];
    const float* g1   = (const float*)d_in[5];
    const float* be1  = (const float*)d_in[6];
    const float* W2   = (const float*)d_in[7];
    const float* b2   = (const float*)d_in[8];
    const float* g2   = (const float*)d_in[9];
    const float* be2  = (const float*)d_in[10];
    float* out = (float*)d_out;

    // Workspace layout (all chunks 256B-aligned by construction)
    char* ws = (char*)d_ws;
    const size_t sz_h0 = (size_t)B_SZ * V_POOL * C_IN  * sizeof(_Float16); // 20.97 MB
    const size_t sz_h1 = (size_t)B_SZ * V_POOL * C_OUT * sizeof(_Float16); // 41.95 MB
    const size_t sz_c  = (size_t)B_SZ * V_POOL * C_OUT * sizeof(float);    // 83.90 MB (reused)
    const size_t sz_w1 = (size_t)7 * C_IN  * C_OUT * sizeof(_Float16);
    const size_t sz_w2 = (size_t)7 * C_OUT * C_OUT * sizeof(_Float16);
    _Float16* h0   = (_Float16*)(ws);
    _Float16* h1   = (_Float16*)(ws + sz_h0);
    float*    cbuf = (float*)   (ws + sz_h0 + sz_h1);
    _Float16* wt1  = (_Float16*)(ws + sz_h0 + sz_h1 + sz_c);
    _Float16* wt2  = (_Float16*)(ws + sz_h0 + sz_h1 + sz_c + sz_w1);
    float*    st   = (float*)   (ws + sz_h0 + sz_h1 + sz_c + sz_w1 + sz_w2);
    float* mean1 = st;
    float* rinv1 = st + (size_t)B_SZ * C_OUT;
    float* mean2 = st + (size_t)2 * B_SZ * C_OUT;
    float* rinv2 = st + (size_t)3 * B_SZ * C_OUT;

    const int elemsBN = B_SZ * V_POOL * C_OUT;              // 20,975,616
    const int gBN     = (elemsBN + 255) / 256;              // 81,936
    const int gPool   = (B_SZ * V_POOL * C_IN + 255) / 256; // 40,968
    const int mTiles  = (B_SZ * V_POOL) / 16;               // 10,242
    const int gGemm   = (mTiles + 7) / 8;                   // 1,281

    // 1) pool + f16 convert
    pool_kernel<<<gPool, 256, 0, stream>>>(x, pidx, h0);
    // 2) weight transpose/convert
    wconv_kernel<<<(7 * C_IN  * C_OUT + 255) / 256, 256, 0, stream>>>(W1, wt1, 7 * C_IN);
    wconv_kernel<<<(7 * C_OUT * C_OUT + 255) / 256, 256, 0, stream>>>(W2, wt2, 7 * C_OUT);
    // 3) conv1 (WMMA)
    gemm_kernel<C_IN><<<gGemm, 256, 0, stream>>>(h0, nidx, wt1, b1, cbuf);
    // 4) BN1 stats + apply -> h1 (f16)
    bnstats_kernel<<<B_SZ * C_OUT, 256, 0, stream>>>(cbuf, mean1, rinv1);
    bnact_f16_kernel<<<gBN, 256, 0, stream>>>(cbuf, mean1, rinv1, g1, be1, h1);
    // 5) conv2 (WMMA) — reuses cbuf
    gemm_kernel<C_OUT><<<gGemm, 256, 0, stream>>>(h1, nidx, wt2, b2, cbuf);
    // 6) BN2 stats + apply + transpose -> out
    bnstats_kernel<<<B_SZ * C_OUT, 256, 0, stream>>>(cbuf, mean2, rinv2);
    bnact_out_kernel<<<gBN, 256, 0, stream>>>(cbuf, mean2, rinv2, g2, be2, out);
}
